// RNNBase_9208409882730
// MI455X (gfx1250) — matrix operand order repaired and checked by reading, not compiled, not run on hardware
//
#include <hip/hip_runtime.h>

// Persistent 2-layer GRU for MI455X (gfx1250, wave32, WMMA).
// grid = 32 batches x 13 row-tiles; block = 256 threads = 8 waves.
// Each block owns 16 (b,n) rows and iterates all T=64 steps locally.
// GEMMs via v_wmma_f32_16x16x32_f16; recurrent state carried in f32 VGPRs;
// h staged through f16 LDS (padded stride -> bank-conflict-free ds_load_b128);
// gate nonlinearities via native v_tanh_f32 when available.

typedef __attribute__((ext_vector_type(16))) _Float16 v16h;
typedef __attribute__((ext_vector_type(8)))  _Float16 v8h;
typedef __attribute__((ext_vector_type(8)))  float    v8f;

namespace {
constexpr int kB = 32, kT = 64, kN = 207, kF = 64, kH = 128, kG = 384; // G = 3H
constexpr int kTilesPerB  = 13;                  // ceil(207/16)
constexpr int kPanelHalfs = kH * 16;             // 2048 halves per [128 x 16] panel
constexpr int kPanels     = kG / 16;             // 24 column panels per matrix
constexpr int kWeightHalfs = 3 * kPanels * kPanelHalfs;  // Whh0, Wih1, Whh1 in LDS
constexpr int kStride     = kH + 8;              // 136-half padded stage row stride
constexpr size_t kSmemBytes = (size_t)kWeightHalfs * 2        // 294912 B f16 weights
                            + (size_t)2 * 16 * kStride * 2;   // 8704 B h0/h1 f16 stage
}

// gfx1250 has a native v_tanh_f32 transcendental; use it for both gates.
#if defined(__gfx1250__) && __has_builtin(__builtin_amdgcn_tanhf)
__device__ __forceinline__ float tanhx(float x) { return __builtin_amdgcn_tanhf(x); }
#else
__device__ __forceinline__ float tanhx(float x) {
  return 1.0f - 2.0f / (__expf(2.0f * x) + 1.0f);
}
#endif
__device__ __forceinline__ float sigm(float x) {
  return 0.5f * tanhx(0.5f * x) + 0.5f;     // sigmoid via tanh: 1 trans op
}

__device__ __forceinline__ v8f wmma16(v16h a, v16h b, v8f c) {
  return __builtin_amdgcn_wmma_f32_16x16x32_f16(false, a, false, b, (short)0, c, false, false);
}

__device__ __forceinline__ v16h combine8(v8h a0, v8h a1) {
  v16h r;
#pragma unroll
  for (int i = 0; i < 8; ++i) { r[i] = a0[i]; r[8 + i] = a1[i]; }
  return r;
}

__device__ __forceinline__ v16h zero16h() {
  v16h r;
#pragma unroll
  for (int i = 0; i < 16; ++i) r[i] = (_Float16)0.0f;
  return r;
}

// B fragment (32x16 f16, KxN) from an LDS panel pre-swizzled per 32-k chunk:
// lane L holds column L%16, 16 K halves in table order = 32 contiguous bytes.
__device__ __forceinline__ v16h ldb(const _Float16* p) {
  return combine8(*(const v8h*)p, *(const v8h*)(p + 8));
}

// A fragment (16x32 f16, MxK) from the f16 h stage: lane holds row M=lane%16;
// halves 0..7 = K chunk at 32f+8*hi, halves 8..15 = chunk at 32f+16+8*hi.
__device__ __forceinline__ v16h ld_a_stage(const _Float16* s, int lo, int f, int hi) {
  const _Float16* p = s + lo * kStride + 32 * f + 8 * hi;
  return combine8(*(const v8h*)p, *(const v8h*)(p + 16));
}

// A fragment from global f32 x rows (convert on the fly; only 2 frags/step).
__device__ __forceinline__ v16h cvt_a(const float* __restrict__ c0,
                                      const float* __restrict__ c1) {
  v16h a;
#pragma unroll
  for (int i = 0; i < 8; ++i) a[i] = (_Float16)c0[i];
#pragma unroll
  for (int i = 0; i < 8; ++i) a[8 + i] = (_Float16)c1[i];
  return a;
}

__device__ __forceinline__ v8f bcast8(float s) {
  v8f v;
#pragma unroll
  for (int i = 0; i < 8; ++i) v[i] = s;
  return v;
}

__global__ __launch_bounds__(256, 1) void gru2_wmma_kernel(
    const float* __restrict__ x,
    const float* __restrict__ Wih0, const float* __restrict__ Whh0,
    const float* __restrict__ bih0, const float* __restrict__ bhh0,
    const float* __restrict__ Wih1, const float* __restrict__ Whh1,
    const float* __restrict__ bih1, const float* __restrict__ bhh1,
    float* __restrict__ out) {
  extern __shared__ char smem[];
  _Float16* sW   = (_Float16*)smem;                      // 3 matrices of 24 panels
  _Float16* sH0h = (_Float16*)(smem + (size_t)kWeightHalfs * 2);
  _Float16* sH1h = sH0h + 16 * kStride;

  const int b    = blockIdx.x / kTilesPerB;
  const int n0   = (blockIdx.x % kTilesPerB) * 16;
  const int tid  = threadIdx.x;
  const int w    = tid >> 5;        // wave id 0..7 -> hidden cols [16w,16w+16)
  const int lane = tid & 31;
  const int hi   = lane >> 4;
  const int lo   = lane & 15;

  // ---- Stage recurrent weights into LDS as f16 column panels (once) ----
  for (int idx = tid; idx < 3 * kH * kG; idx += 256) {
    int m = idx / (kH * kG);
    int r = idx % (kH * kG);
    int k = r / kG, n = r % kG;
    const float* Wm = (m == 0) ? Whh0 : ((m == 1) ? Wih1 : Whh1);
    sW[(m * kPanels + (n >> 4)) * kPanelHalfs + (k >> 5) * 512 + ((((k >> 3) & 1) * 16 + (n & 15)) * 16) + (k & 7) + ((k >> 4) & 1) * 8] = (_Float16)Wm[r];
  }
  for (int idx = tid; idx < 16 * kStride; idx += 256) {
    sH0h[idx] = (_Float16)0.0f;
    sH1h[idx] = (_Float16)0.0f;
  }

  // ---- Per-wave register-resident W_ih0 fragments (K=64 -> 2 frags/gate) ----
  v16h Bih[3][2];
#pragma unroll
  for (int g = 0; g < 3; ++g) {
    const int ncol = g * kH + 16 * w;
#pragma unroll
    for (int f = 0; f < 2; ++f) {
      const float* p = Wih0 + (size_t)(32 * f + 8 * hi) * kG + ncol + lo;
      v16h v;
#pragma unroll
      for (int h = 0; h < 16; ++h) v[h] = (_Float16)p[(size_t)(h < 8 ? h : h + 8) * kG];
      Bih[g][f] = v;
    }
  }

  const int cr = 16 * w + lo, cz = kH + cr, cn = 2 * kH + cr;
  const float b0r = bih0[cr] + bhh0[cr], b0z = bih0[cz] + bhh0[cz];
  const float b0i = bih0[cn],            b0h = bhh0[cn];
  const float b1r = bih1[cr] + bhh1[cr], b1z = bih1[cz] + bhh1[cz];
  const float b1i = bih1[cn],            b1h = bhh1[cn];

  const _Float16* Phh0 = sW;
  const _Float16* Pih1 = sW + (size_t)kPanels * kPanelHalfs;
  const _Float16* Phh1 = sW + (size_t)2 * kPanels * kPanelHalfs;
  const _Float16* p0r = Phh0 + (0 * 8 + w) * kPanelHalfs;
  const _Float16* p0z = Phh0 + (1 * 8 + w) * kPanelHalfs;
  const _Float16* p0n = Phh0 + (2 * 8 + w) * kPanelHalfs;
  const _Float16* p1r = Pih1 + (0 * 8 + w) * kPanelHalfs;
  const _Float16* p1z = Pih1 + (1 * 8 + w) * kPanelHalfs;
  const _Float16* p1n = Pih1 + (2 * 8 + w) * kPanelHalfs;
  const _Float16* p2r = Phh1 + (0 * 8 + w) * kPanelHalfs;
  const _Float16* p2z = Phh1 + (1 * 8 + w) * kPanelHalfs;
  const _Float16* p2n = Phh1 + (2 * 8 + w) * kPanelHalfs;

  // Recurrent state in f32 registers, WMMA C-layout: acc[i] = (row i+8*hi, col 16w+lo)
  v8f h0 = bcast8(0.0f), h1 = bcast8(0.0f);

  // h0 A-fragments kept live across the loop: the fragments staged for layer 1
  // at step t are exactly the layer-0 fragments at step t+1. h0(0)=0 -> zeros.
  v16h Ah[4];
#pragma unroll
  for (int f = 0; f < 4; ++f) Ah[f] = zero16h();

  const int  arow  = (n0 + lo < kN) ? (n0 + lo) : (kN - 1);  // clamp tail rows
  const float* xbase = x + ((size_t)b * kT * kN + arow) * kF;

  __syncthreads();

#pragma unroll 1
  for (int t = 0; t < kT; ++t) {
    // ================= layer 0 =================
    const float* xr = xbase + (size_t)t * kN * kF;
    v16h Ax[2];
#pragma unroll
    for (int f = 0; f < 2; ++f)
      Ax[f] = cvt_a(xr + 32 * f + 8 * hi, xr + 32 * f + 16 + 8 * hi);
    if (t + 1 < kT) {                       // global_prefetch next timestep's rows
      __builtin_prefetch(xr + (size_t)kN * kF + 8 * hi, 0, 3);
      __builtin_prefetch(xr + (size_t)kN * kF + 32 + 8 * hi, 0, 3);
    }

    v8f ar = bcast8(b0r), az = bcast8(b0z), ai = bcast8(b0i), ah = bcast8(b0h);
#pragma unroll
    for (int f = 0; f < 2; ++f) {
      ar = wmma16(Ax[f], Bih[0][f], ar);
      az = wmma16(Ax[f], Bih[1][f], az);
      ai = wmma16(Ax[f], Bih[2][f], ai);
    }
#pragma unroll
    for (int f = 0; f < 4; ++f) {
      ar = wmma16(Ah[f], ldb(p0r + (32 * f + lane) * 16), ar);
      az = wmma16(Ah[f], ldb(p0z + (32 * f + lane) * 16), az);
      ah = wmma16(Ah[f], ldb(p0n + (32 * f + lane) * 16), ah);
    }
#pragma unroll
    for (int i = 0; i < 8; ++i) {          // gate math fully in registers
      float r = sigm(ar[i]);
      float z = sigm(az[i]);
      float n = tanhx(ai[i] + r * ah[i]);
      h0[i] = (1.0f - z) * n + z * h0[i];
    }
    __syncthreads();                       // all waves done reading old sH0h
#pragma unroll
    for (int i = 0; i < 8; ++i)
      sH0h[(i + 8 * hi) * kStride + 16 * w + lo] = (_Float16)h0[i];
    __syncthreads();                       // new h0 staged

    // ================= layer 1 =================
    v16h A1[4];
#pragma unroll
    for (int f = 0; f < 4; ++f) {
      Ah[f] = ld_a_stage(sH0h, lo, f, hi); // h0n: used now AND by next layer 0
      A1[f] = ld_a_stage(sH1h, lo, f, hi); // old h1
    }
    ar = bcast8(b1r); az = bcast8(b1z); ai = bcast8(b1i); ah = bcast8(b1h);
#pragma unroll
    for (int f = 0; f < 4; ++f) {
      ar = wmma16(Ah[f], ldb(p1r + (32 * f + lane) * 16), ar);
      ar = wmma16(A1[f], ldb(p2r + (32 * f + lane) * 16), ar);
      az = wmma16(Ah[f], ldb(p1z + (32 * f + lane) * 16), az);
      az = wmma16(A1[f], ldb(p2z + (32 * f + lane) * 16), az);
      ai = wmma16(Ah[f], ldb(p1n + (32 * f + lane) * 16), ai);
      ah = wmma16(A1[f], ldb(p2n + (32 * f + lane) * 16), ah);
    }
#pragma unroll
    for (int i = 0; i < 8; ++i) {
      float r = sigm(ar[i]);
      float z = sigm(az[i]);
      float n = tanhx(ai[i] + r * ah[i]);
      h1[i] = (1.0f - z) * n + z * h1[i];
    }
    __syncthreads();                       // all waves done reading old sH1h
#pragma unroll
    for (int i = 0; i < 8; ++i)
      sH1h[(i + 8 * hi) * kStride + 16 * w + lo] = (_Float16)h1[i];

    // emit last layer's state for this timestep
#pragma unroll
    for (int i = 0; i < 8; ++i) {
      int n = n0 + i + 8 * hi;
      if (n < kN)
        out[(((size_t)b * kT + t) * kN + n) * kH + 16 * w + lo] = h1[i];
    }
  }

  // final hidden states: d_out = [out | h0f | h1f]
  float* h0f = out + (size_t)kB * kT * kN * kH;
  float* h1f = h0f + (size_t)kB * kN * kH;
#pragma unroll
  for (int i = 0; i < 8; ++i) {
    int n = n0 + i + 8 * hi;
    if (n < kN) {
      size_t o = ((size_t)b * kN + n) * kH + 16 * w + lo;
      h0f[o] = h0[i];
      h1f[o] = h1[i];
    }
  }
}

extern "C" void kernel_launch(void* const* d_in, const int* in_sizes, int n_in,
                              void* d_out, int out_size, void* d_ws, size_t ws_size,
                              hipStream_t stream) {
  (void)in_sizes; (void)n_in; (void)out_size; (void)d_ws; (void)ws_size;
  const float* x    = (const float*)d_in[0];
  const float* Wih0 = (const float*)d_in[1];
  const float* Whh0 = (const float*)d_in[2];
  const float* bih0 = (const float*)d_in[3];
  const float* bhh0 = (const float*)d_in[4];
  const float* Wih1 = (const float*)d_in[5];
  const float* Whh1 = (const float*)d_in[6];
  const float* bih1 = (const float*)d_in[7];
  const float* bhh1 = (const float*)d_in[8];
  float* out = (float*)d_out;

  dim3 grid(kB * kTilesPerB);   // 416 blocks: one per (b, 16-row tile)
  gru2_wmma_kernel<<<grid, 256, kSmemBytes, stream>>>(
      x, Wih0, Whh0, bih0, bhh0, Wih1, Whh1, bih1, bhh1, out);
}
